// PixelSNAIL_48601849921553
// MI455X (gfx1250) — matrix operand order repaired and checked by reading, not compiled, not run on hardware
//
#include <hip/hip_runtime.h>
#include <hip/hip_bf16.h>

typedef __attribute__((ext_vector_type(16))) _Float16 v16h;
typedef __attribute__((ext_vector_type(8)))  _Float16 v8h;
typedef __attribute__((ext_vector_type(8)))  float    v8f;

#define NPOS 4096   // 4 * 32 * 32

// -------- tap tables: mtype 0 = mask A (12 taps), 1 = mask B (13 taps), 2 = 1x1 center --------
__device__ __forceinline__ void tap_off(int mtype, int t, int& dy, int& dx) {
  if (mtype == 2) { dy = 0; dx = 0; return; }
  if (t < 10)      { dy = (t / 5) - 2; dx = (t % 5) - 2; }   // rows -2,-1 all cols
  else if (t < 12) { dy = 0;           dx = t - 12;      }   // row 0, cols -2,-1
  else             { dy = 0;           dx = 0;           }   // center (mask B only)
}

// ---------------------------------------------------------------------------
__global__ void embed_kernel(const float* __restrict__ emb, const int* __restrict__ xi,
                             float* __restrict__ cur) {
  long long i = (long long)blockIdx.x * 256 + threadIdx.x;
  if (i >= (long long)NPOS * 256) return;
  int p = (int)(i & 1023);
  long long bc = i >> 10;
  int c = (int)(bc & 255);
  int b = (int)(bc >> 8);
  int code = xi[(b << 10) | p];
  cur[i] = emb[(size_t)code * 256 + c];
}

// pack NCHW fp32 slab (C channels) into transposed f16 XT[pos][c0+c] (optional elu)
__global__ void pack_kernel(const float* __restrict__ src, int C, int c0, int stride, int doelu,
                            _Float16* __restrict__ XT) {
  long long i = (long long)blockIdx.x * 256 + threadIdx.x;
  long long tot = (long long)NPOS * C;
  if (i >= tot) return;
  int c = (int)(i % C);
  int n = (int)(i / C);
  int b = n >> 10, p = n & 1023;
  float v = src[((size_t)b * C + c) * 1024 + p];
  if (doelu) v = v > 0.f ? v : (__expf(v) - 1.f);
  XT[(size_t)n * stride + c0 + c] = (_Float16)v;
}

__global__ void pospack_kernel(int c0, int stride, _Float16* __restrict__ XT) {
  int n = blockIdx.x * 256 + threadIdx.x;
  if (n >= NPOS) return;
  int p = n & 1023;
  int y = p >> 5, x = p & 31;
  XT[(size_t)n * stride + c0]     = (_Float16)((float)y * (1.f / 32.f) - 0.5f);
  XT[(size_t)n * stride + c0 + 1] = (_Float16)((float)x * (1.f / 32.f) - 0.5f);
}

__global__ void wzero_kernel(_Float16* __restrict__ p, long long nelem) {
  long long i = (long long)blockIdx.x * 256 + threadIdx.x;
  if (i < nelem) p[i] = (_Float16)0.f;
}

// pack fp32 conv weight [Cout][CinSrc][KK][KK] into f16 Wp[tap][Cout][Kpad] slice at c0
__global__ void wpack_kernel(const float* __restrict__ Wsrc, int CinSrc, int KK, int mtype,
                             int ntaps, int Cout, int Kpad, int c0, _Float16* __restrict__ Wp) {
  long long i = (long long)blockIdx.x * 256 + threadIdx.x;
  long long tot = (long long)ntaps * Cout * CinSrc;
  if (i >= tot) return;
  int ci = (int)(i % CinSrc);
  long long t2 = i / CinSrc;
  int co = (int)(t2 % Cout);
  int t  = (int)(t2 / Cout);
  int dy, dx; tap_off(mtype, t, dy, dx);
  int kh = KK >> 1;
  int ky = dy + kh, kx = dx + kh;
  Wp[((size_t)t * Cout + co) * Kpad + c0 + ci] =
      (_Float16)Wsrc[(((size_t)co * CinSrc + ci) * KK + ky) * KK + kx];
}

// ---------------------------------------------------------------------------
// Generic implicit-GEMM conv via WMMA. One wave computes a 16(Cout) x 16(pos) tile.
// epi 0: store fp32 NCHW (out32). epi 1: elu + store f16 transposed (outT).
__global__ void conv_wmma_kernel(const _Float16* __restrict__ Wp, const _Float16* __restrict__ XT,
                                 int xts, const float* __restrict__ bias,
                                 const float* __restrict__ bias2,
                                 int Cout, int Kpad, int ntaps, int mtype, int epi,
                                 float* __restrict__ out32, _Float16* __restrict__ outT, int oTs) {
  union AF { v16h v; v8h h[2]; };
  const int lane = threadIdx.x & 31;
  const int half = lane >> 4;
  const int mr   = lane & 15;
  const int n0  = blockIdx.x << 4;
  const int co0 = blockIdx.y << 4;
  const int n = n0 + mr;
  const int bimg = n >> 10;
  const int p = n & 1023;
  const int y = p >> 5, x = p & 31;

  v8f acc; for (int i = 0; i < 8; ++i) acc[i] = 0.f;
  v8h hz;  for (int i = 0; i < 8; ++i) hz[i] = (_Float16)0.f;

  for (int t = 0; t < ntaps; ++t) {
    int dy, dx; tap_off(mtype, t, dy, dx);
    const int sy = y + dy, sx = x + dx;
    const bool valid = (sy >= 0) && (sy < 32) && (sx >= 0) && (sx < 32);
    const _Float16* ar = Wp + ((size_t)t * Cout + (co0 + mr)) * Kpad;
    const _Float16* br = valid ? (XT + (size_t)(bimg * 1024 + sy * 32 + sx) * xts) : nullptr;
    for (int k0 = 0; k0 < Kpad; k0 += 32) {
      AF a, bf;
      a.h[0] = *(const v8h*)(ar + k0 + half * 8);
      a.h[1] = *(const v8h*)(ar + k0 + 16 + half * 8);
      if (valid) {
        bf.h[0] = *(const v8h*)(br + k0 + half * 8);
        bf.h[1] = *(const v8h*)(br + k0 + 16 + half * 8);
      } else { bf.h[0] = hz; bf.h[1] = hz; }
      acc = __builtin_amdgcn_wmma_f32_16x16x32_f16(false, a.v, false, bf.v,
                                                   (short)0, acc, false, false);
    }
  }
#pragma unroll
  for (int v = 0; v < 8; ++v) {
    const int m = co0 + v + half * 8;
    float val = acc[v] + bias[m] + (bias2 ? bias2[m] : 0.f);
    if (epi == 0) {
      out32[((size_t)(bimg * Cout + m) << 10) + p] = val;
    } else {
      float e = val > 0.f ? val : (__expf(val) - 1.f);
      outT[(size_t)n * oTs + m] = (_Float16)e;
    }
  }
}

// conv (Cout=512) fused with GLU + residual: cur = a*sigmoid(g) + cur. Shares B fragment
// between the two 16x16 output tiles (channels m and m+256).
__global__ void conv_glu_kernel(const _Float16* __restrict__ Wp, const _Float16* __restrict__ XT,
                                int xts, const float* __restrict__ bias,
                                int Kpad, int ntaps, int mtype, float* __restrict__ cur) {
  union AF { v16h v; v8h h[2]; };
  const int lane = threadIdx.x & 31;
  const int half = lane >> 4;
  const int mr   = lane & 15;
  const int n0  = blockIdx.x << 4;
  const int co0 = blockIdx.y << 4;
  const int n = n0 + mr;
  const int bimg = n >> 10;
  const int p = n & 1023;
  const int y = p >> 5, x = p & 31;

  v8f acca, accg;
  for (int i = 0; i < 8; ++i) { acca[i] = 0.f; accg[i] = 0.f; }
  v8h hz; for (int i = 0; i < 8; ++i) hz[i] = (_Float16)0.f;

  for (int t = 0; t < ntaps; ++t) {
    int dy, dx; tap_off(mtype, t, dy, dx);
    const int sy = y + dy, sx = x + dx;
    const bool valid = (sy >= 0) && (sy < 32) && (sx >= 0) && (sx < 32);
    const _Float16* ara = Wp + ((size_t)t * 512 + (co0 + mr)) * Kpad;
    const _Float16* arg = Wp + ((size_t)t * 512 + 256 + (co0 + mr)) * Kpad;
    const _Float16* br  = valid ? (XT + (size_t)(bimg * 1024 + sy * 32 + sx) * xts) : nullptr;
    for (int k0 = 0; k0 < Kpad; k0 += 32) {
      AF a1, a2, bf;
      a1.h[0] = *(const v8h*)(ara + k0 + half * 8);
      a1.h[1] = *(const v8h*)(ara + k0 + 16 + half * 8);
      a2.h[0] = *(const v8h*)(arg + k0 + half * 8);
      a2.h[1] = *(const v8h*)(arg + k0 + 16 + half * 8);
      if (valid) {
        bf.h[0] = *(const v8h*)(br + k0 + half * 8);
        bf.h[1] = *(const v8h*)(br + k0 + 16 + half * 8);
      } else { bf.h[0] = hz; bf.h[1] = hz; }
      acca = __builtin_amdgcn_wmma_f32_16x16x32_f16(false, a1.v, false, bf.v,
                                                    (short)0, acca, false, false);
      accg = __builtin_amdgcn_wmma_f32_16x16x32_f16(false, a2.v, false, bf.v,
                                                    (short)0, accg, false, false);
    }
  }
#pragma unroll
  for (int v = 0; v < 8; ++v) {
    const int m = co0 + v + half * 8;
    float av = acca[v] + bias[m];
    float gv = accg[v] + bias[256 + m];
    float sg = 1.f / (1.f + __expf(-gv));
    size_t oi = ((size_t)(bimg * 256 + m) << 10) + p;
    cur[oi] = av * sg + cur[oi];
  }
}

// LayerNorm over channels + ELU, in place. One wave per position.
__global__ void ln_elu_kernel(float* __restrict__ cur, const float* __restrict__ g,
                              const float* __restrict__ bt) {
  const int wave = threadIdx.x >> 5;
  const int lane = threadIdx.x & 31;
  const int pos = blockIdx.x * 8 + wave;
  const int b = pos >> 10, p = pos & 1023;
  float v[8];
  float s1 = 0.f, s2 = 0.f;
#pragma unroll
  for (int i = 0; i < 8; ++i) {
    int c = lane + 32 * i;
    float xv = cur[((size_t)(b * 256 + c)) * 1024 + p];
    v[i] = xv; s1 += xv; s2 += xv * xv;
  }
  for (int off = 16; off; off >>= 1) {
    s1 += __shfl_xor(s1, off);
    s2 += __shfl_xor(s2, off);
  }
  float mu  = s1 * (1.f / 256.f);
  float var = s2 * (1.f / 256.f) - mu * mu;
  float rs  = rsqrtf(var + 1e-5f);
#pragma unroll
  for (int i = 0; i < 8; ++i) {
    int c = lane + 32 * i;
    float yv = (v[i] - mu) * rs * g[c] + bt[c];
    yv = yv > 0.f ? yv : (__expf(yv) - 1.f);
    cur[((size_t)(b * 256 + c)) * 1024 + p] = yv;
  }
}

// Causal multi-head attention (d=8, n=1024). One block per (batch, head); K/V in LDS.
__global__ void attn_kernel(const float* __restrict__ qbuf, const float* __restrict__ kvbuf,
                            float* __restrict__ abuf) {
  __shared__ float ksh[1024 * 8];
  __shared__ float vsh[1024 * 8];
  const int bh = blockIdx.x;
  const int b = bh >> 3, hd = bh & 7;
  const int tid = threadIdx.x;
  for (int i = tid; i < 8192; i += 256) {
    int d = i & 7, kp = i >> 3;
    ksh[i] = kvbuf[((size_t)(b * 128 + hd * 8 + d)) * 1024 + kp];
    vsh[i] = kvbuf[((size_t)(b * 128 + 64 + hd * 8 + d)) * 1024 + kp];
  }
  __syncthreads();
  const float scale = 0.35355339059327373f;  // 1/sqrt(8)
  for (int qp = tid; qp < 1024; qp += 256) {
    float qd[8];
#pragma unroll
    for (int d = 0; d < 8; ++d)
      qd[d] = qbuf[((size_t)(b * 64 + hd * 8 + d)) * 1024 + qp] * scale;
    float m = -1e30f, s = 0.f;
    float o[8]; for (int d = 0; d < 8; ++d) o[d] = 0.f;
    for (int kp = 0; kp <= qp; ++kp) {
      float l = 0.f;
#pragma unroll
      for (int d = 0; d < 8; ++d) l += qd[d] * ksh[kp * 8 + d];
      float nm = l > m ? l : m;
      float corr = __expf(m - nm);
      float e = __expf(l - nm);
      s = s * corr + e;
#pragma unroll
      for (int d = 0; d < 8; ++d) o[d] = o[d] * corr + e * vsh[kp * 8 + d];
      m = nm;
    }
    float inv = 1.f / s;
#pragma unroll
    for (int d = 0; d < 8; ++d)
      abuf[((size_t)(b * 64 + hd * 8 + d)) * 1024 + qp] = o[d] * inv;
  }
}

// ---------------------------------------------------------------------------
extern "C" void kernel_launch(void* const* d_in, const int* in_sizes, int n_in,
                              void* d_out, int out_size, void* d_ws, size_t ws_size,
                              hipStream_t stream) {
  const float* F[64];
  for (int i = 0; i < n_in && i < 64; ++i) F[i] = (const float*)d_in[i];

  // Index maps: insertion order (x first, 4096 elems) vs pytree-sorted order (kv_b first).
  int I_x, I_embed, I_in_w, I_in_b, I_or_w1, I_or_b1, I_or_w2, I_or_b2;
  int I_flng, I_flnb, I_fw, I_fb;
  int QW[2], QB[2], KVW[2], KVB[2], OW1[2], OB1[2], OW2[2], OB2[2], AW[2], AB[2];
  int RW1[2][2], RB1[2][2], RW2[2][2], RB2[2][2], RLG[2][2], RLB[2][2];

  if (in_sizes[0] == 4096) {            // insertion order
    I_x = 0; I_embed = 1; I_in_w = 2; I_in_b = 3;
    int i = 4;
    for (int b = 0; b < 2; ++b) {
      for (int r = 0; r < 2; ++r) {
        RW1[b][r] = i++; RB1[b][r] = i++; RW2[b][r] = i++; RB2[b][r] = i++;
        RLG[b][r] = i++; RLB[b][r] = i++;
      }
      QW[b] = i++; QB[b] = i++; KVW[b] = i++; KVB[b] = i++;
      OW1[b] = i++; OB1[b] = i++; OW2[b] = i++; OB2[b] = i++; AW[b] = i++; AB[b] = i++;
    }
    I_or_w1 = i++; I_or_b1 = i++; I_or_w2 = i++; I_or_b2 = i++;
    I_flng = i++; I_flnb = i++; I_fw = i++; I_fb = i++;
  } else {                               // sorted (pytree) order
    int i = 0;
    for (int b = 0; b < 2; ++b) {
      KVB[b] = i++; KVW[b] = i++;
      AB[b] = i++; AW[b] = i++; OB1[b] = i++; OB2[b] = i++; OW1[b] = i++; OW2[b] = i++;
      QB[b] = i++; QW[b] = i++;
      for (int r = 0; r < 2; ++r) {
        RB1[b][r] = i++; RB2[b][r] = i++; RLB[b][r] = i++; RLG[b][r] = i++;
        RW1[b][r] = i++; RW2[b][r] = i++;
      }
    }
    I_embed = i++; I_fb = i++; I_flnb = i++; I_flng = i++; I_fw = i++;
    I_in_b = i++; I_in_w = i++;
    I_or_b1 = i++; I_or_b2 = i++; I_or_w1 = i++; I_or_w2 = i++;
    I_x = i++;
  }

  // ---- workspace carve ----
  char* w = (char*)d_ws;
  float* cur  = (float*)w;      w += (size_t)NPOS * 256 * 4;   // stream NCHW
  float* xblk = (float*)w;      w += (size_t)NPOS * 256 * 4;   // block input copy
  float* qb   = (float*)w;      w += (size_t)NPOS * 64 * 4;
  float* kvb  = (float*)w;      w += (size_t)NPOS * 128 * 4;
  float* ab   = (float*)w;      w += (size_t)NPOS * 64 * 4;
  _Float16* XT  = (_Float16*)w; w += (size_t)NPOS * 544 * 2;   // transposed activations
  _Float16* XTm = (_Float16*)w; w += (size_t)NPOS * 256 * 2;   // gated-res mid
  _Float16* Wp  = (_Float16*)w; w += (size_t)13 * 512 * 256 * 2;
  (void)ws_size;

  hipStream_t st = stream;

  auto packA = [&](const float* src, int C, int c0, int stride, int doelu) {
    long long nn = (long long)NPOS * C;
    pack_kernel<<<(unsigned)((nn + 255) / 256), 256, 0, st>>>(src, C, c0, stride, doelu, XT);
  };
  auto conv = [&](const float* Wsrc, int CinSrc, int KK, int mtype, int ntaps,
                  int Cout, int Kpad, const float* b1, const float* b2,
                  int xts, int epi, float* o32, _Float16* oT, int oTs) {
    long long wn = (long long)ntaps * Cout * Kpad;
    wzero_kernel<<<(unsigned)((wn + 255) / 256), 256, 0, st>>>(Wp, wn);
    long long pn = (long long)ntaps * Cout * CinSrc;
    wpack_kernel<<<(unsigned)((pn + 255) / 256), 256, 0, st>>>(Wsrc, CinSrc, KK, mtype,
                                                               ntaps, Cout, Kpad, 0, Wp);
    conv_wmma_kernel<<<dim3(NPOS / 16, Cout / 16), 32, 0, st>>>(Wp, XT, xts, b1, b2,
                                                                Cout, Kpad, ntaps, mtype,
                                                                epi, o32, oT, oTs);
  };
  auto glu = [&](const float* Wsrc, int KK, int mtype, int ntaps, const float* b2) {
    long long wn = (long long)ntaps * 512 * 256;
    wzero_kernel<<<(unsigned)((wn + 255) / 256), 256, 0, st>>>(Wp, wn);
    long long pn = (long long)ntaps * 512 * 256;
    wpack_kernel<<<(unsigned)((pn + 255) / 256), 256, 0, st>>>(Wsrc, 256, KK, mtype,
                                                               ntaps, 512, 256, 0, Wp);
    conv_glu_kernel<<<dim3(NPOS / 16, 16), 32, 0, st>>>(Wp, XTm, 256, b2, 256, ntaps, mtype, cur);
  };

  // ---- embedding + input masked conv (mask A, 12 taps) ----
  {
    long long nn = (long long)NPOS * 256;
    embed_kernel<<<(unsigned)((nn + 255) / 256), 256, 0, st>>>(F[I_embed],
                                                               (const int*)d_in[I_x], cur);
  }
  packA(cur, 256, 0, 256, 0);
  conv(F[I_in_w], 256, 5, 0, 12, 256, 256, F[I_in_b], nullptr, 256, 0, cur, nullptr, 0);

  // ---- blocks ----
  for (int b = 0; b < 2; ++b) {
    hipMemcpyAsync(xblk, cur, (size_t)NPOS * 256 * 4, hipMemcpyDeviceToDevice, st);
    for (int r = 0; r < 2; ++r) {
      packA(cur, 256, 0, 256, 0);
      conv(F[RW1[b][r]], 256, 5, 1, 13, 256, 256, F[RB1[b][r]], nullptr,
           256, 1, nullptr, XTm, 256);                       // elu(conv1) -> XTm
      glu(F[RW2[b][r]], 5, 1, 13, F[RB2[b][r]]);             // cur = glu(conv2)+cur
      ln_elu_kernel<<<NPOS / 8, 256, 0, st>>>(cur, F[RLG[b][r]], F[RLB[b][r]]);
    }
    // attention: q from [out|pos] (258ch), kv from [out|pos|x_blk] (514ch)
    packA(cur, 256, 0, 544, 0);
    pospack_kernel<<<NPOS / 256, 256, 0, st>>>(256, 544, XT);
    packA(xblk, 256, 258, 544, 0);
    conv(F[QW[b]], 258, 1, 2, 1, 64, 288, F[QB[b]], nullptr, 544, 0, qb, nullptr, 0);
    conv(F[KVW[b]], 514, 1, 2, 1, 128, 544, F[KVB[b]], nullptr, 544, 0, kvb, nullptr, 0);
    attn_kernel<<<32, 256, 0, st>>>(qb, kvb, ab);
    // out gated-res with aux: conv over concat[out | elu(a)] (320ch), bias = b1 + aux_b
    packA(cur, 256, 0, 320, 0);
    packA(ab, 64, 256, 320, 1);
    {
      long long wn = 256LL * 320;
      wzero_kernel<<<(unsigned)((wn + 255) / 256), 256, 0, st>>>(Wp, wn);
      long long p1 = 256LL * 256;
      wpack_kernel<<<(unsigned)((p1 + 255) / 256), 256, 0, st>>>(F[OW1[b]], 256, 1, 2, 1,
                                                                 256, 320, 0, Wp);
      long long p2 = 256LL * 64;
      wpack_kernel<<<(unsigned)((p2 + 255) / 256), 256, 0, st>>>(F[AW[b]], 64, 1, 2, 1,
                                                                 256, 320, 256, Wp);
      conv_wmma_kernel<<<dim3(NPOS / 16, 16), 32, 0, st>>>(Wp, XT, 320, F[OB1[b]], F[AB[b]],
                                                           256, 320, 1, 2, 1,
                                                           nullptr, XTm, 256);
    }
    glu(F[OW2[b]], 1, 2, 1, F[OB2[b]]);
  }

  // ---- out_res (1x1 gated-res, no LN inside) ----
  packA(cur, 256, 0, 256, 0);
  conv(F[I_or_w1], 256, 1, 2, 1, 256, 256, F[I_or_b1], nullptr, 256, 1, nullptr, XTm, 256);
  glu(F[I_or_w2], 1, 2, 1, F[I_or_b2]);

  // ---- final LN + ELU + 1x1 conv to 512 logits ----
  ln_elu_kernel<<<NPOS / 8, 256, 0, st>>>(cur, F[I_flng], F[I_flnb]);
  packA(cur, 256, 0, 256, 0);
  conv(F[I_fw], 256, 1, 2, 1, 512, 256, F[I_fb], nullptr, 256, 0, (float*)d_out, nullptr, 0);
  (void)out_size; (void)n_in;
}
